// DecoderOperator_8804682957031
// MI455X (gfx1250) — matrix operand order repaired and checked by reading, not compiled, not run on hardware
//
#include <hip/hip_runtime.h>
#include <hip/hip_bf16.h>

// ---------------- problem constants ----------------
#define NB   8
#define INC  2
#define HCC  64
#define OUTC 2
#define HH   128
#define WW   128
#define KTAPS 13   // masked 5x5: rows 0,1 full + (2,0),(2,1),(2,2); type A zeroes tap 12

typedef __bf16 bf16;
typedef __attribute__((ext_vector_type(16))) __bf16 v16bf;
typedef __attribute__((ext_vector_type(8)))  __bf16 v8bf;
typedef __attribute__((ext_vector_type(8)))  float  v8f;

union V16U { v16bf v; unsigned short u[16]; };

__device__ __forceinline__ unsigned short f2bf(float f) {
    unsigned int u = __float_as_uint(f);
    u += 0x7fffu + ((u >> 16) & 1u);          // round-to-nearest-even
    return (unsigned short)(u >> 16);
}
__device__ __forceinline__ float bf2f(unsigned short s) {
    return __uint_as_float(((unsigned int)s) << 16);
}

__device__ __forceinline__ float sigmoidf_(float x) { return 1.f / (1.f + __expf(-x)); }
__device__ __forceinline__ float mishf_(float x) {
    float sp = (x > 15.f) ? x : log1pf(__expf(x));
    return x * tanhf(sp);
}

enum { MODE_MISH = 0, MODE_PRE = 1, MODE_MISH_ADD = 2, MODE_FINAL = 3 };

// ---------------- weight packing ----------------
struct Ptr7 { const float* p[7]; };

// w1pack[s][t][oc][cin] bf16 (tap-major -> each tap slice is 16KB contiguous),
// mask applied; set 1 (block0) is type 'A' -> tap 12 zeroed
__global__ void pack_w1_kernel(Ptr7 w1s, unsigned short* w1p) {
    int idx = blockIdx.x * 256 + threadIdx.x;
    const int TOT = 7 * KTAPS * 128 * 64;
    if (idx >= TOT) return;
    int cin = idx & 63;
    int oc  = (idx >> 6) & 127;
    int t   = (idx / (64 * 128)) % KTAPS;
    int s   = idx / (64 * 128 * KTAPS);
    const int tkh[KTAPS] = {0,0,0,0,0,1,1,1,1,1,2,2,2};
    const int tkw[KTAPS] = {0,1,2,3,4,0,1,2,3,4,0,1,2};
    float w = w1s.p[s][((size_t)(oc * 64 + cin) * 5 + tkh[t]) * 5 + tkw[t]];
    if (s == 1 && t == 12) w = 0.f;   // type 'A' excludes center pixel
    w1p[idx] = f2bf(w);
}

// w2pack[s][oc2(0..63)][hc] bf16; rows beyond real out_c zero-padded
__global__ void pack_w2_kernel(Ptr7 w2s, unsigned short* w2p) {
    int idx = blockIdx.x * 256 + threadIdx.x;
    const int TOT = 7 * 64 * 64;
    if (idx >= TOT) return;
    int hc = idx & 63;
    int oc = (idx >> 6) & 63;
    int s  = idx >> 12;
    int outc = (s == 6) ? OUTC : 64;
    float w = (oc < outc) ? w2s.p[s][oc * 64 + hc] : 0.f;
    w2p[idx] = f2bf(w);
}

// ebf[b,h,r,i] = bf16(e);  extT[b,h,i,r] = bf16(e+x)   (both from NCHW fp32)
__global__ void prep_e_kernel(const float* __restrict__ e, const float* __restrict__ x,
                              unsigned short* __restrict__ ebf, unsigned short* __restrict__ extT) {
    int idx = blockIdx.x * 256 + threadIdx.x;
    const int TOT = NB * HCC * HH * WW;
    if (idx >= TOT) return;
    ebf[idx] = f2bf(e[idx]);
    int r  = idx & 127;
    int i  = (idx >> 7) & 127;
    int bh = idx >> 14;
    size_t src = ((size_t)bh << 14) + ((size_t)r << 7) + i;
    extT[idx] = f2bf(e[src] + x[src]);
}

// ---------------- causal gated conv (C_in=2, type 'A'), fp32 VALU ----------------
// out r: NHWC bf16 [B,H,W,64]
__global__ __launch_bounds__(256) void causal_gated_kernel(
        const float* __restrict__ yin,   // NCHW [8,2,128,128]
        const float* __restrict__ W1c, const float* __restrict__ b1c,
        const float* __restrict__ W2c, const float* __restrict__ b2c,
        unsigned short* __restrict__ rout) {
    __shared__ float glds[4][64];
    int tid = threadIdx.x;
    int p = tid >> 6, hc = tid & 63;
    int bid = blockIdx.x;
    int xb = bid & 31; int by = bid >> 5;
    int y = by & 127;  int b = by >> 7;
    int x = xb * 4 + p;

    const int tkh[12] = {0,0,0,0,0,1,1,1,1,1,2,2};
    const int tkw[12] = {0,1,2,3,4,0,1,2,3,4,0,1};
    float av = b1c[hc], gv = b1c[64 + hc];
    for (int t = 0; t < 12; t++) {
        int yy = y + tkh[t] - 2, xx = x + tkw[t] - 2;
        if (yy < 0 || yy >= HH || xx < 0 || xx >= WW) continue;
        for (int cin = 0; cin < 2; cin++) {
            float xv = yin[(((size_t)b * 2 + cin) * HH + yy) * WW + xx];
            av += xv * W1c[((size_t)(hc * 2 + cin) * 5 + tkh[t]) * 5 + tkw[t]];
            gv += xv * W1c[((size_t)((hc + 64) * 2 + cin) * 5 + tkh[t]) * 5 + tkw[t]];
        }
    }
    glds[p][hc] = tanhf(av) * sigmoidf_(gv);
    __syncthreads();
    float v = b2c[hc];
    for (int k = 0; k < 64; k++) v += W2c[hc * 64 + k] * glds[p][k];
    rout[(((size_t)b * HH + y) * WW + x) * 64 + hc] = f2bf(v);
}

// ---------------- main gated conv: implicit-GEMM WMMA ----------------
// input NHWC bf16 [B,H,W,64]; 4 waves/block, each wave = 16 pixels of one row.
// Per-tap weights (16KB) are double-buffered through LDS with a software pipeline:
//   issue tap t+1 global loads -> compute tap t from LDS -> drain regs to LDS -> barrier.
template <int MODE>
__global__ __launch_bounds__(128) void gconv_wmma_kernel(
        const unsigned short* __restrict__ in,    // NHWC bf16
        const unsigned short* __restrict__ w1p,   // [13][128][64] bf16 (tap-major)
        const float* __restrict__ b1,
        const unsigned short* __restrict__ w2p,   // [64][64] bf16
        const float* __restrict__ b2,
        const unsigned short* __restrict__ aux,   // r (PRE) or conv input (MISH_ADD), NHWC bf16
        unsigned short* __restrict__ outw,        // NHWC bf16 (MISH/MISH_ADD) or NCHW bf16 (PRE)
        float* __restrict__ outf) {               // NCHW f32 (FINAL)
    __shared__ __align__(16) unsigned short wbuf[2][128 * 64];   // 2 x 16KB weight stage
    __shared__ __align__(16) unsigned short glds[4][16][64];     // gated activations
    int tid = threadIdx.x;
    int wave = tid >> 5, lane = tid & 31;
    int m = lane & 15, hi = lane >> 4;
    int bid = blockIdx.x;
    int xblk = bid & 1; int by = bid >> 1;
    int y = by & 127;   int b = by >> 7;
    int x0 = xblk * 64 + wave * 16;
    int px = x0 + m;

    const int tkh[KTAPS] = {0,0,0,0,0,1,1,1,1,1,2,2,2};
    const int tkw[KTAPS] = {0,1,2,3,4,0,1,2,3,4,0,1,2};

    v8f acc[8];
#pragma unroll
    for (int n = 0; n < 8; n++) acc[n] = (v8f){0.f,0.f,0.f,0.f,0.f,0.f,0.f,0.f};

    // each thread stages one oc-row (64 bf16 = 128B = 8 x uint4) per tap
    uint4 st[8];
    {
        const uint4* src = (const uint4*)(w1p + (size_t)tid * 64);   // tap 0
#pragma unroll
        for (int j = 0; j < 8; j++) st[j] = src[j];
        uint4* dst = (uint4*)&wbuf[0][tid * 64];
#pragma unroll
        for (int j = 0; j < 8; j++) dst[j] = st[j];
    }
    __syncthreads();

    for (int t = 0; t < KTAPS; t++) {
        // 1) issue next tap's weight loads early (latency hidden by this tap's WMMAs)
        if (t + 1 < KTAPS) {
            const uint4* src = (const uint4*)(w1p + (size_t)(t + 1) * 128 * 64 + tid * 64);
#pragma unroll
            for (int j = 0; j < 8; j++) st[j] = src[j];
        }

        // 2) A fragments for both K-chunks of this tap (global, bf16 NHWC)
        int yy = y + tkh[t] - 2;
        int xx = px + tkw[t] - 2;
        bool ok = (yy >= 0) && (yy < HH) && (xx >= 0) && (xx < WW);
        V16U a0, a1;
        if (ok) {
            const unsigned short* p = in + (((size_t)b * HH + yy) * WW + xx) * 64 + hi * 16;
            *(v8bf*)&a0.u[0] = *(const v8bf*)(p);        // K {hi*16..+7}
            *(v8bf*)&a0.u[8] = *(const v8bf*)(p + 8);    // K {hi*16+8..+15}
            *(v8bf*)&a1.u[0] = *(const v8bf*)(p + 32);   // K {32+hi*16..+7}
            *(v8bf*)&a1.u[8] = *(const v8bf*)(p + 40);   // K {32+hi*16+8..+15}
        } else {
#pragma unroll
            for (int j = 0; j < 16; j++) { a0.u[j] = 0; a1.u[j] = 0; }
        }

        // 3) 16 WMMAs from LDS-resident weights
        const unsigned short* wb = wbuf[t & 1];
#pragma unroll
        for (int n = 0; n < 8; n++) {
            int oc = n * 16 + m;
            v16bf b0 = *(const v16bf*)&wb[oc * 64 + hi * 16];
            v16bf b1v = *(const v16bf*)&wb[oc * 64 + 32 + hi * 16];
            acc[n] = __builtin_amdgcn_wmma_f32_16x16x32_bf16(
                         false, a0.v, false, b0, (short)0, acc[n], false, false);
            acc[n] = __builtin_amdgcn_wmma_f32_16x16x32_bf16(
                         false, a1.v, false, b1v, (short)0, acc[n], false, false);
        }

        // 4) drain staged regs into the other LDS buffer
        if (t + 1 < KTAPS) {
            uint4* dst = (uint4*)&wbuf[(t + 1) & 1][tid * 64];
#pragma unroll
            for (int j = 0; j < 8; j++) dst[j] = st[j];
        }
        __syncthreads();
    }

    // ---- bias + gate: gated = tanh(a)*sigmoid(g); stage to LDS as bf16 ----
#pragma unroll
    for (int n = 0; n < 4; n++) {
        float b1a = b1[n * 16 + m];
        float b1g = b1[64 + n * 16 + m];
#pragma unroll
        for (int rv = 0; rv < 8; rv++) {
            float aval = acc[n][rv] + b1a;
            float gval = acc[n + 4][rv] + b1g;
            int pix = rv + hi * 8;
            glds[wave][pix][n * 16 + m] = f2bf(tanhf(aval) * sigmoidf_(gval));
        }
    }
    __syncthreads();

    // ---- 1x1 conv: M=16 pixels, K=64 hc, N = out_c ----
    constexpr int NT2 = (MODE == MODE_FINAL) ? 1 : 4;
    v8f acc2[NT2];
#pragma unroll
    for (int n = 0; n < NT2; n++) acc2[n] = (v8f){0.f,0.f,0.f,0.f,0.f,0.f,0.f,0.f};
#pragma unroll
    for (int c0 = 0; c0 < 64; c0 += 32) {
        V16U a;
        *(v8bf*)&a.u[0] = *(const v8bf*)&glds[wave][m][c0 + hi * 16];
        *(v8bf*)&a.u[8] = *(const v8bf*)&glds[wave][m][c0 + hi * 16 + 8];
#pragma unroll
        for (int n2 = 0; n2 < NT2; n2++) {
            int oc2 = n2 * 16 + m;
            const unsigned short* wp = w2p + (size_t)oc2 * 64 + c0 + hi * 16;
            v16bf bb = *(const v16bf*)wp;
            acc2[n2] = __builtin_amdgcn_wmma_f32_16x16x32_bf16(
                           false, a.v, false, bb, (short)0, acc2[n2], false, false);
        }
    }

    // ---- epilogue ----
#pragma unroll
    for (int n2 = 0; n2 < NT2; n2++) {
        int oc2 = n2 * 16 + m;
        float bias2 = (MODE == MODE_FINAL) ? ((oc2 < OUTC) ? b2[oc2] : 0.f) : b2[oc2];
#pragma unroll
        for (int rv = 0; rv < 8; rv++) {
            float v = acc2[n2][rv] + bias2;
            int pix = rv + hi * 8;
            int xg = x0 + pix;
            size_t nhwc = (((size_t)b * HH + y) * WW + xg) * 64 + oc2;
            if (MODE == MODE_MISH) {
                outw[nhwc] = f2bf(mishf_(v));
            } else if (MODE == MODE_PRE) {
                float d = v + bf2f(aux[nhwc]);
                outw[(((size_t)b * 64 + oc2) * HH + y) * WW + xg] = f2bf(d);
            } else if (MODE == MODE_MISH_ADD) {
                outw[nhwc] = f2bf(mishf_(v) + bf2f(aux[nhwc]));
            } else { // FINAL: NCHW fp32, out_c = 2
                if (oc2 < OUTC)
                    outf[(((size_t)b * OUTC + oc2) * HH + y) * WW + xg] = v;
            }
        }
    }
}

// ---------------- attention scores: S = D * E^T, quadrant mask ----------------
__global__ __launch_bounds__(32) void attn_scores_kernel(
        const unsigned short* __restrict__ d,    // [b,h,t,i] bf16
        const unsigned short* __restrict__ ebf,  // [b,h,r,i] bf16
        float* __restrict__ sbuf) {              // [b,h,t,r] f32
    int lane = threadIdx.x;
    int m = lane & 15, hi = lane >> 4;
    int bid = blockIdx.x;
    int tt = bid & 7; int bh = bid >> 3;
    int t0 = tt * 16;
    const unsigned short* drow = d + (size_t)bh * HH * WW;
    const unsigned short* erow = ebf + (size_t)bh * HH * WW;

    v8f acc[8];
#pragma unroll
    for (int n = 0; n < 8; n++) acc[n] = (v8f){0.f,0.f,0.f,0.f,0.f,0.f,0.f,0.f};

#pragma unroll
    for (int c0 = 0; c0 < 128; c0 += 32) {
        V16U a;
        const unsigned short* ap = drow + (size_t)(t0 + m) * 128 + c0 + hi * 16;
        *(v8bf*)&a.u[0] = *(const v8bf*)(ap);
        *(v8bf*)&a.u[8] = *(const v8bf*)(ap + 8);
#pragma unroll
        for (int n = 0; n < 8; n++) {
            const unsigned short* bp = erow + (size_t)(n * 16 + m) * 128 + c0 + hi * 16;
            v16bf bb = *(const v16bf*)bp;
            acc[n] = __builtin_amdgcn_wmma_f32_16x16x32_bf16(
                         false, a.v, false, bb, (short)0, acc[n], false, false);
        }
    }
#pragma unroll
    for (int n = 0; n < 8; n++) {
        int r = n * 16 + m;
#pragma unroll
        for (int rv = 0; rv < 8; rv++) {
            int t = t0 + rv + hi * 8;
            float v = acc[n][rv];
            if (t >= 65 && r >= 65) v = 0.f;   // mask lower-right quadrant (score -> 0)
            sbuf[(size_t)bh * 16384 + (size_t)t * 128 + r] = v;
        }
    }
}

// ---------------- softmax over r (masked zeros participate, per reference) ----------------
__global__ __launch_bounds__(32) void softmax_kernel(
        const float* __restrict__ sbuf, unsigned short* __restrict__ pbuf) {
    int row = blockIdx.x;                    // bh*128 + t
    int lane = threadIdx.x;
    const float* p = sbuf + (size_t)row * 128;
    float v[4];
    float mx = -1e30f;
#pragma unroll
    for (int j = 0; j < 4; j++) { v[j] = p[lane + 32 * j]; mx = fmaxf(mx, v[j]); }
#pragma unroll
    for (int o = 16; o >= 1; o >>= 1) mx = fmaxf(mx, __shfl_xor(mx, o, 32));
    float s = 0.f;
#pragma unroll
    for (int j = 0; j < 4; j++) { v[j] = __expf(v[j] - mx); s += v[j]; }
#pragma unroll
    for (int o = 16; o >= 1; o >>= 1) s += __shfl_xor(s, o, 32);
    float inv = 1.f / s;
#pragma unroll
    for (int j = 0; j < 4; j++) pbuf[(size_t)row * 128 + lane + 32 * j] = f2bf(v[j] * inv);
}

// ---------------- attention output: h2 = h + P * (E+X) ----------------
__global__ __launch_bounds__(32) void attn_out_kernel(
        const unsigned short* __restrict__ pbuf,   // [b,h,t,r] bf16
        const unsigned short* __restrict__ extT,   // [b,h,i,r] bf16
        const unsigned short* __restrict__ hin,    // NHWC bf16
        unsigned short* __restrict__ hout) {       // NHWC bf16
    int lane = threadIdx.x;
    int m = lane & 15, hi = lane >> 4;
    int bid = blockIdx.x;
    int tt = bid & 7; int bh = bid >> 3;
    int t0 = tt * 16;
    int b = bh >> 6, h = bh & 63;
    const unsigned short* prow = pbuf + (size_t)bh * 16384;
    const unsigned short* erow = extT + (size_t)bh * 16384;

    v8f acc[8];
#pragma unroll
    for (int n = 0; n < 8; n++) acc[n] = (v8f){0.f,0.f,0.f,0.f,0.f,0.f,0.f,0.f};

#pragma unroll
    for (int c0 = 0; c0 < 128; c0 += 32) {      // K = r
        V16U a;
        const unsigned short* ap = prow + (size_t)(t0 + m) * 128 + c0 + hi * 16;
        *(v8bf*)&a.u[0] = *(const v8bf*)(ap);
        *(v8bf*)&a.u[8] = *(const v8bf*)(ap + 8);
#pragma unroll
        for (int n = 0; n < 8; n++) {           // N = i
            const unsigned short* bp = erow + (size_t)(n * 16 + m) * 128 + c0 + hi * 16;
            v16bf bb = *(const v16bf*)bp;
            acc[n] = __builtin_amdgcn_wmma_f32_16x16x32_bf16(
                         false, a.v, false, bb, (short)0, acc[n], false, false);
        }
    }
#pragma unroll
    for (int n = 0; n < 8; n++) {
        int i = n * 16 + m;
#pragma unroll
        for (int rv = 0; rv < 8; rv++) {
            int t = t0 + rv + hi * 8;
            size_t o = (((size_t)b * HH + t) * WW + i) * 64 + h;
            hout[o] = f2bf(bf2f(hin[o]) + acc[n][rv]);
        }
    }
}

// ---------------- host-side orchestration ----------------
extern "C" void kernel_launch(void* const* d_in, const int* in_sizes, int n_in,
                              void* d_out, int out_size, void* d_ws, size_t ws_size,
                              hipStream_t stream) {
    (void)in_sizes; (void)n_in; (void)out_size; (void)ws_size;

    // inputs (setup_inputs insertion order, params flattened depth-first)
    const float* yin = (const float*)d_in[0];
    const float* xin = (const float*)d_in[1];
    const float* ein = (const float*)d_in[2];
    const float* cW1 = (const float*)d_in[3];
    const float* cb1 = (const float*)d_in[4];
    const float* cW2 = (const float*)d_in[5];
    const float* cb2 = (const float*)d_in[6];
    // set s: 0 = pre, 1..6 = blocks[0..5]
    Ptr7 w1s, w2s;
    const float* b1s[7]; const float* b2s[7];
    w1s.p[0] = (const float*)d_in[7];  b1s[0] = (const float*)d_in[8];
    w2s.p[0] = (const float*)d_in[9];  b2s[0] = (const float*)d_in[10];
    for (int i = 0; i < 6; i++) {
        w1s.p[1 + i] = (const float*)d_in[11 + 4 * i];
        b1s[1 + i]   = (const float*)d_in[12 + 4 * i];
        w2s.p[1 + i] = (const float*)d_in[13 + 4 * i];
        b2s[1 + i]   = (const float*)d_in[14 + 4 * i];
    }

    // workspace layout
    char* ws = (char*)d_ws;
    size_t off = 0;
    const size_t ACT_F32 = (size_t)NB * HH * WW * 64 * sizeof(float);   // 33.5 MB
    const size_t ACT_BF  = (size_t)NB * 64 * HH * WW * sizeof(short);   // 16.8 MB
    unsigned short* rbuf = (unsigned short*)(ws + off); off += ACT_BF;  // NHWC bf16
    unsigned short* hA   = (unsigned short*)(ws + off); off += ACT_BF;  // NHWC bf16
    unsigned short* hB   = (unsigned short*)(ws + off); off += ACT_BF;  // NHWC bf16
    float*          sbuf = (float*)(ws + off);          off += ACT_F32; // scores f32
    unsigned short* dbuf = (unsigned short*)(ws + off); off += ACT_BF;  // NCHW bf16
    unsigned short* pbuf = (unsigned short*)(ws + off); off += ACT_BF;
    unsigned short* ebf  = (unsigned short*)(ws + off); off += ACT_BF;
    unsigned short* extT = (unsigned short*)(ws + off); off += ACT_BF;
    unsigned short* w1p  = (unsigned short*)(ws + off); off += (size_t)7 * KTAPS * 128 * 64 * 2;
    unsigned short* w2p  = (unsigned short*)(ws + off); off += (size_t)7 * 64 * 64 * 2;

    // 1) pack weights + e/(e+x)^T
    pack_w1_kernel<<<(7 * KTAPS * 128 * 64 + 255) / 256, 256, 0, stream>>>(w1s, w1p);
    pack_w2_kernel<<<(7 * 64 * 64 + 255) / 256, 256, 0, stream>>>(w2s, w2p);
    prep_e_kernel<<<(NB * HCC * HH * WW + 255) / 256, 256, 0, stream>>>(ein, xin, ebf, extT);

    // 2) r = gated_conv(y, causal, 'A')  (cheap, VALU)
    causal_gated_kernel<<<NB * HH * (WW / 4), 256, 0, stream>>>(yin, cW1, cb1, cW2, cb2, rbuf);

    const int GC_GRID = NB * HH * (WW / 64);   // 2048 blocks x 128 threads
    auto w1set = [&](int s) { return w1p + (size_t)s * KTAPS * 128 * 64; };
    auto w2set = [&](int s) { return w2p + (size_t)s * 64 * 64; };

    // 3) i=0: h = mish(gconv(r, blocks[0], 'A'))
    gconv_wmma_kernel<MODE_MISH><<<GC_GRID, 128, 0, stream>>>(
        rbuf, w1set(1), b1s[1], w2set(1), b2s[1], nullptr, hA, nullptr);

    // 4) i=1..4
    for (int i = 1; i <= 4; i++) {
        // d = gconv(h, pre, 'B') + r   -> bf16 NCHW
        gconv_wmma_kernel<MODE_PRE><<<GC_GRID, 128, 0, stream>>>(
            hA, w1set(0), b1s[0], w2set(0), b2s[0], rbuf, dbuf, nullptr);
        // attention
        attn_scores_kernel<<<NB * HCC * (HH / 16), 32, 0, stream>>>(dbuf, ebf, sbuf);
        softmax_kernel<<<NB * HCC * HH, 32, 0, stream>>>(sbuf, pbuf);
        attn_out_kernel<<<NB * HCC * (HH / 16), 32, 0, stream>>>(pbuf, extT, hA, hB);
        // h = mish(gconv(h2, blocks[i], 'B')) + h2
        gconv_wmma_kernel<MODE_MISH_ADD><<<GC_GRID, 128, 0, stream>>>(
            hB, w1set(1 + i), b1s[1 + i], w2set(1 + i), b2s[1 + i], hB, hA, nullptr);
    }

    // 5) i=5: y_tilde = gconv(h, blocks[5], 'B')  -> NCHW fp32 d_out
    gconv_wmma_kernel<MODE_FINAL><<<GC_GRID, 128, 0, stream>>>(
        hA, w1set(6), b1s[6], w2set(6), b2s[6], nullptr, nullptr, (float*)d_out);
}
